// GraniteMoeMoE_43963285242550
// MI455X (gfx1250) — matrix operand order repaired and checked by reading, not compiled, not run on hardware
//
#include <hip/hip_runtime.h>
#include <hip/hip_bf16.h>

// Problem constants (GraniteMoeMoE): B=2,S=1024,D=2048,F=1024,E=8,K=2
#define TOK    2048   // B*S tokens
#define DIM    2048   // D (model dim)
#define FF     1024   // F (ffn dim)
#define NEXP   8      // E experts
#define TILE_M 32     // tokens per expert tile (two WMMA M-subtiles)

typedef __attribute__((ext_vector_type(16))) __bf16 bf16x16;
typedef __attribute__((ext_vector_type(8)))  float  f32x8;

static_assert(sizeof(bf16x16) == 32, "bf16x16 must be 8 VGPRs");

// ---------------------------------------------------------------------------
// WMMA helpers: D = A(16x32 bf16) x B(32x16 bf16) + C(16x16 f32)
// A layout: lane m(0-15)/m+16 holds row m; lane<16 -> K[0..15],
// lane>=16 -> K[16..31]  (one contiguous 32B chunk, same k order as B).
// B layout: lane n(0-15)/n+16 holds column n = weight row n; lane<16 ->
// K[0..15], lane>=16 -> K[16..31] (one contiguous 32B chunk).
// C/D layout: VGPR r, lane l -> (M = r + 8*(l>>4), N = l&15).
// ---------------------------------------------------------------------------
__device__ __forceinline__ f32x8 wmma_bf16(bf16x16 a, bf16x16 b, f32x8 c) {
  return __builtin_amdgcn_wmma_f32_16x16x32_bf16(
      /*neg_a=*/false, a, /*neg_b=*/false, b,
      /*c_mod=*/(short)0, c, /*reuse_a=*/false, /*reuse_b=*/false);
}

__device__ __forceinline__ bf16x16 load_a_frag(const __hip_bfloat16* row,
                                               int k0, int hi) {
  bf16x16 r;
  ((uint4*)&r)[0] = *(const uint4*)(row + k0 + hi * 16);
  ((uint4*)&r)[1] = *(const uint4*)(row + k0 + hi * 16 + 8);
  return r;
}

__device__ __forceinline__ bf16x16 load_b_frag(const __hip_bfloat16* p) {
  bf16x16 r;
  ((uint4*)&r)[0] = ((const uint4*)p)[0];
  ((uint4*)&r)[1] = ((const uint4*)p)[1];
  return r;
}

// ---------------------------------------------------------------------------
// Kernel 1: zero the output accumulator and per-expert counters.
// ---------------------------------------------------------------------------
__global__ void zero_kernel(float* __restrict__ out, int n4,
                            int* __restrict__ counts) {
  int i = blockIdx.x * blockDim.x + threadIdx.x;
  const int stride = gridDim.x * blockDim.x;
  const float4 z = {0.f, 0.f, 0.f, 0.f};
  for (; i < n4; i += stride) ((float4*)out)[i] = z;
  if (blockIdx.x == 0 && threadIdx.x < NEXP) counts[threadIdx.x] = 0;
}

// ---------------------------------------------------------------------------
// Kernel 2: fp32 -> bf16 conversion (vectorized 4-wide).
// ---------------------------------------------------------------------------
__global__ void cvt_bf16_kernel(const float* __restrict__ in,
                                __hip_bfloat16* __restrict__ out, int n4) {
  int i = blockIdx.x * blockDim.x + threadIdx.x;
  const int stride = gridDim.x * blockDim.x;
  for (; i < n4; i += stride) {
    float4 v = ((const float4*)in)[i];
    union { __hip_bfloat16 h[4]; uint2 u; } pk;
    pk.h[0] = __float2bfloat16(v.x);
    pk.h[1] = __float2bfloat16(v.y);
    pk.h[2] = __float2bfloat16(v.z);
    pk.h[3] = __float2bfloat16(v.w);
    ((uint2*)out)[i] = pk.u;
  }
}

// ---------------------------------------------------------------------------
// Kernel 3: router. One 256-thread block (8 waves) per token: 8 logits of
// length-2048 dot products in fp32, wave shfl_xor reduction, top-2 + softmax,
// atomic append (token, gate) to the owning experts' lists.
// ---------------------------------------------------------------------------
__global__ __launch_bounds__(256) void router_kernel(
    const float* __restrict__ x, const float* __restrict__ wr,
    int* __restrict__ counts, int* __restrict__ tok_list,
    float* __restrict__ gate_list) {
  const int t = blockIdx.x;
  const int tid = threadIdx.x;
  const float* xr = x + (size_t)t * DIM;

  float p[NEXP];
#pragma unroll
  for (int e = 0; e < NEXP; ++e) p[e] = 0.f;

  for (int d = tid; d < DIM; d += 256) {
    const float xv = xr[d];
#pragma unroll
    for (int e = 0; e < NEXP; ++e) p[e] += xv * wr[e * DIM + d];
  }
  // wave32 reduction
#pragma unroll
  for (int m = 16; m >= 1; m >>= 1) {
#pragma unroll
    for (int e = 0; e < NEXP; ++e) p[e] += __shfl_xor(p[e], m, 32);
  }
  __shared__ float sred[NEXP][8];
  if ((tid & 31) == 0) {
#pragma unroll
    for (int e = 0; e < NEXP; ++e) sred[e][tid >> 5] = p[e];
  }
  __syncthreads();
  if (tid == 0) {
    float l[NEXP];
#pragma unroll
    for (int e = 0; e < NEXP; ++e) {
      float s = 0.f;
#pragma unroll
      for (int w = 0; w < 8; ++w) s += sred[e][w];
      l[e] = s;
    }
    int i1 = 0;
#pragma unroll
    for (int e = 1; e < NEXP; ++e)
      if (l[e] > l[i1]) i1 = e;
    int i2 = (i1 == 0) ? 1 : 0;
#pragma unroll
    for (int e = 0; e < NEXP; ++e)
      if (e != i1 && l[e] > l[i2]) i2 = e;
    // softmax over the two selected logits (l[i1] >= l[i2])
    const float e2 = __expf(l[i2] - l[i1]);
    const float inv = 1.f / (1.f + e2);
    int p1 = atomicAdd(&counts[i1], 1);
    tok_list[i1 * TOK + p1] = t;
    gate_list[i1 * TOK + p1] = inv;
    int p2 = atomicAdd(&counts[i2], 1);
    tok_list[i2 * TOK + p2] = t;
    gate_list[i2 * TOK + p2] = e2 * inv;
  }
}

// ---------------------------------------------------------------------------
// Kernel 4: fused expert MLP. grid = (TOK/TILE_M tiles, NEXP experts),
// block = 256 threads = 8 waves. Two 16-row M-subtiles per block so each B
// fragment (weights) feeds 2 WMMAs and each A fragment feeds 2 WMMAs.
// Stage 1: h[32,F] = silu(Xt*Wg^T) * (Xt*Wu^T)  -> LDS (bf16, padded rows)
// Stage 2: y[32,D] = h * Wd^T, scaled by gate, atomic-added into out.
// All WMMA bf16 with f32 accumulation; weights stream from the 192MB L2.
// ---------------------------------------------------------------------------
__global__ __launch_bounds__(256) void moe_kernel(
    const __hip_bfloat16* __restrict__ xb,
    const __hip_bfloat16* __restrict__ wgb,
    const __hip_bfloat16* __restrict__ wub,
    const __hip_bfloat16* __restrict__ wdb,
    const int* __restrict__ counts, const int* __restrict__ tok_list,
    const float* __restrict__ gate_list, float* __restrict__ out) {
  const int e = blockIdx.y;
  const int tile = blockIdx.x;
  const int cnt = counts[e];
  if (tile * TILE_M >= cnt) return;

  __shared__ __hip_bfloat16 sh_h[TILE_M][FF + 8];  // +8 bf16 pad: conflict-free
  __shared__ int sh_tok[TILE_M];
  __shared__ float sh_gate[TILE_M];

  const int tid = threadIdx.x;
  const int wave = tid >> 5;
  const int lane = tid & 31;
  const int ln = lane & 15;
  const int hi = lane >> 4;

  // Async-copy the tile's token ids + gate weights into LDS (ASYNCcnt path).
  if (tid < TILE_M) {
    int idx = tile * TILE_M + tid;
    if (idx > cnt - 1) idx = cnt - 1;  // clamp padded rows to a valid entry
    const int* gp_t = tok_list + (size_t)e * TOK + idx;
    const float* gp_g = gate_list + (size_t)e * TOK + idx;
    unsigned lds_t = (unsigned)(size_t)&sh_tok[tid];
    unsigned lds_g = (unsigned)(size_t)&sh_gate[tid];
    asm volatile("global_load_async_to_lds_b32 %0, %1, off" ::"v"(lds_t),
                 "v"(gp_t)
                 : "memory");
    asm volatile("global_load_async_to_lds_b32 %0, %1, off" ::"v"(lds_g),
                 "v"(gp_g)
                 : "memory");
  }
  asm volatile("s_wait_asynccnt 0x0" ::: "memory");
  __syncthreads();
  if (tid < TILE_M && (tile * TILE_M + tid) >= cnt) sh_gate[tid] = 0.f;
  __syncthreads();

  // Per-lane A-row pointers: row M = lane&15 of each 16-token M-subtile.
  const __hip_bfloat16* arow0 = xb + (size_t)sh_tok[ln] * DIM;
  const __hip_bfloat16* arow1 = xb + (size_t)sh_tok[16 + ln] * DIM;

  // ---- Stage 1: gate/up projections + SiLU, h tile into LDS ----
  for (int ft = wave; ft < FF / 16; ft += 8) {
    const int f0 = ft * 16;
    f32x8 accg0 = {}, accu0 = {}, accg1 = {}, accu1 = {};
    const __hip_bfloat16* bg = wgb + ((size_t)e * FF + f0 + ln) * DIM + hi * 16;
    const __hip_bfloat16* bu = wub + ((size_t)e * FF + f0 + ln) * DIM + hi * 16;
#pragma unroll 2
    for (int k0 = 0; k0 < DIM; k0 += 32) {
      bf16x16 a0 = load_a_frag(arow0, k0, hi);
      bf16x16 a1 = load_a_frag(arow1, k0, hi);
      bf16x16 vg = load_b_frag(bg + k0);
      bf16x16 vu = load_b_frag(bu + k0);
      accg0 = wmma_bf16(a0, vg, accg0);
      accu0 = wmma_bf16(a0, vu, accu0);
      accg1 = wmma_bf16(a1, vg, accg1);
      accu1 = wmma_bf16(a1, vu, accu1);
    }
#pragma unroll
    for (int r = 0; r < 8; ++r) {
      const float g0 = accg0[r], u0 = accu0[r];
      const float g1 = accg1[r], u1 = accu1[r];
      // silu(g)*u with fast HW reciprocal (v_rcp_f32) instead of IEEE divide
      const float h0 = g0 * __builtin_amdgcn_rcpf(1.f + __expf(-g0)) * u0;
      const float h1 = g1 * __builtin_amdgcn_rcpf(1.f + __expf(-g1)) * u1;
      sh_h[r + 8 * hi][f0 + ln] = __float2bfloat16(h0);
      sh_h[16 + r + 8 * hi][f0 + ln] = __float2bfloat16(h1);
    }
  }
  __syncthreads();

  // ---- Stage 2: down projection from LDS h tile, scatter-add to out ----
  const __hip_bfloat16* hrow0 = &sh_h[ln][0];
  const __hip_bfloat16* hrow1 = &sh_h[16 + ln][0];
  for (int dt = wave; dt < DIM / 16; dt += 8) {
    const int d0 = dt * 16;
    f32x8 acc0 = {}, acc1 = {};
    const __hip_bfloat16* bd = wdb + ((size_t)e * DIM + d0 + ln) * FF + hi * 16;
#pragma unroll 2
    for (int k0 = 0; k0 < FF; k0 += 32) {
      bf16x16 a0 = load_a_frag(hrow0, k0, hi);  // ds_load from LDS
      bf16x16 a1 = load_a_frag(hrow1, k0, hi);
      bf16x16 b = load_b_frag(bd + k0);
      acc0 = wmma_bf16(a0, b, acc0);
      acc1 = wmma_bf16(a1, b, acc1);
    }
#pragma unroll
    for (int r = 0; r < 8; ++r) {
      const int m0 = r + 8 * hi;
      const int m1 = 16 + m0;
      const float v0 = acc0[r] * sh_gate[m0];
      const float v1 = acc1[r] * sh_gate[m1];
      atomicAdd(&out[(size_t)sh_tok[m0] * DIM + d0 + ln], v0);
      atomicAdd(&out[(size_t)sh_tok[m1] * DIM + d0 + ln], v1);
    }
  }
}

// ---------------------------------------------------------------------------
// Host launcher
// ---------------------------------------------------------------------------
extern "C" void kernel_launch(void* const* d_in, const int* in_sizes, int n_in,
                              void* d_out, int out_size, void* d_ws,
                              size_t ws_size, hipStream_t stream) {
  (void)in_sizes;
  (void)n_in;
  (void)out_size;
  (void)ws_size;
  const float* x = (const float*)d_in[0];   // [B,S,D]
  const float* wr = (const float*)d_in[1];  // [E,D]
  const float* wg = (const float*)d_in[2];  // [E,F,D]
  const float* wu = (const float*)d_in[3];  // [E,F,D]
  const float* wd = (const float*)d_in[4];  // [E,D,F]
  float* out = (float*)d_out;               // [B,S,D]

  char* ws = (char*)d_ws;
  size_t off = 0;
  auto take = [&](size_t bytes) -> char* {
    char* p = ws + off;
    off += (bytes + 255) & ~(size_t)255;
    return p;
  };
  __hip_bfloat16* xb = (__hip_bfloat16*)take((size_t)TOK * DIM * 2);
  __hip_bfloat16* wgb = (__hip_bfloat16*)take((size_t)NEXP * FF * DIM * 2);
  __hip_bfloat16* wub = (__hip_bfloat16*)take((size_t)NEXP * FF * DIM * 2);
  __hip_bfloat16* wdb = (__hip_bfloat16*)take((size_t)NEXP * DIM * FF * 2);
  int* counts = (int*)take(NEXP * sizeof(int));
  int* tok_list = (int*)take((size_t)NEXP * TOK * sizeof(int));
  float* gate_list = (float*)take((size_t)NEXP * TOK * sizeof(float));

  zero_kernel<<<2048, 256, 0, stream>>>(out, TOK * DIM / 4, counts);
  cvt_bf16_kernel<<<2048, 256, 0, stream>>>(x, xb, TOK * DIM / 4);
  cvt_bf16_kernel<<<4096, 256, 0, stream>>>(wg, wgb, NEXP * FF * DIM / 4);
  cvt_bf16_kernel<<<4096, 256, 0, stream>>>(wu, wub, NEXP * FF * DIM / 4);
  cvt_bf16_kernel<<<4096, 256, 0, stream>>>(wd, wdb, NEXP * DIM * FF / 4);
  router_kernel<<<TOK, 256, 0, stream>>>(x, wr, counts, tok_list, gate_list);
  moe_kernel<<<dim3(TOK / TILE_M, NEXP), 256, 0, stream>>>(
      xb, wgb, wub, wdb, counts, tok_list, gate_list, out);
}
